// dense_warp_34565896798773
// MI455X (gfx1250) — matrix operand order repaired and checked by reading, not compiled, not run on hardware
//
#include <hip/hip_runtime.h>

#define B_ 4
#define C_ 128
#define H_ 256
#define W_ 512
#define D_ 64

#define WBLK 128     // w-span per workgroup
#define NWT  8       // 16-wide output tiles per block
#define SLAB_W 208   // per-wave h1 slab width: (NWT-1)*16 + 96 = 208 (16B-multiple rows)
#define CT_W 112     // s_costT row: 16 zero | 64 cost | 32 zero

typedef __attribute__((ext_vector_type(16))) __bf16          v16bf;
typedef __attribute__((ext_vector_type(16))) unsigned short  v16u;
typedef __attribute__((ext_vector_type(8)))  float           v8f;

__device__ __forceinline__ unsigned short f2bf(float f) {
  union { float f; unsigned u; } v; v.f = f;
  unsigned r = v.u + 0x7FFFu + ((v.u >> 16) & 1u);  // round-to-nearest-even
  return (unsigned short)(r >> 16);
}

__global__ __launch_bounds__(256)
void dense_warp_34565896798773_kernel(const float* __restrict__ h1,
                                      const float* __restrict__ cost,
                                      float* __restrict__ out) {
  // Banded Kmat, transposed + zero-padded: s_costT[w][16 + (k - w)]
  __shared__ __align__(16) unsigned short s_costT[WBLK][CT_W];   // 28672 B
  __shared__ __align__(16) unsigned short s_h1[8][16][SLAB_W];   // 53248 B

  const int t    = threadIdx.x;
  const int lane = t & 31;
  const int wv   = t >> 5;
  const int bh   = blockIdx.x;
  const int b    = bh / H_;
  const int h    = bh % H_;
  const int wb   = blockIdx.y * WBLK;

  // ---- zero the padded cost band (16B stores) ----
  {
    uint4 z4; z4.x = z4.y = z4.z = z4.w = 0u;
    uint4* ctz = (uint4*)&s_costT[0][0];
    for (int i = t; i < (WBLK * CT_W * 2) / 16; i += 256) ctz[i] = z4;
  }
  __syncthreads();

  // ---- fill cost band: float4 coalesced reads, transposed u16 LDS writes ----
  {
    const float4* cb4 = (const float4*)(cost + ((size_t)(b * D_) * H_ + h) * W_ + wb);
    const size_t drow4 = (size_t)(H_ * W_) / 4;
    for (int i = t; i < (D_ * WBLK) / 4; i += 256) {
      int d  = i >> 5;          // WBLK/4 == 32
      int w4 = i & 31;
      float4 v = cb4[(size_t)d * drow4 + w4];
      int w = 4 * w4;
      s_costT[w + 0][16 + d] = f2bf(v.x);
      s_costT[w + 1][16 + d] = f2bf(v.y);
      s_costT[w + 2][16 + d] = f2bf(v.z);
      s_costT[w + 3][16 + d] = f2bf(v.w);
    }
  }

  // ---- per-wave h1 slab: rows c0..c0+15, cols wb..wb+207 (zero past W) ----
  const int c0 = wv * 16;
  {
    const float* hrow0 = h1 + (((size_t)(b * C_ + c0)) * H_ + h) * W_;
    for (int r = 0; r < 16; ++r) {
      const float* hr = hrow0 + (size_t)r * (H_ * W_);
      unsigned short* sr = &s_h1[wv][r][0];
      for (int j4 = lane; j4 < SLAB_W / 4; j4 += 32) {
        int w = wb + 4 * j4;
        uint2 p;
        if (w < W_) {                       // W is a multiple of 4: all-or-nothing
          float4 v = *(const float4*)(hr + w);
          p.x = (unsigned)f2bf(v.x) | ((unsigned)f2bf(v.y) << 16);
          p.y = (unsigned)f2bf(v.z) | ((unsigned)f2bf(v.w) << 16);
        } else {
          p.x = 0u; p.y = 0u;
        }
        *(uint2*)(sr + 4 * j4) = p;
      }
    }
  }
  __syncthreads();

  const int m  = lane & 15;   // A row / B,C column within tile
  const int hi = lane >> 4;   // half-wave K-group select
  float* obase = out + (((size_t)(b * C_ + c0)) * H_ + h) * W_;

  for (int wt = 0; wt < NWT; ++wt) {
    const int w0l = wt * 16;
    v8f acc = {0.f, 0.f, 0.f, 0.f, 0.f, 0.f, 0.f, 0.f};

#pragma unroll
    for (int kc = 0; kc < 3; ++kc) {        // K = 96 covers band of 80
      const int kk0 = kc * 32;
      v16u au, bu;
#pragma unroll
      for (int j = 0; j < 8; ++j) {
        // A 16x32 bf16 layout: VGPR j<4 -> K = 8*hi + 2j (+1)
        //                      VGPR j>=4 -> K = 16 + 8*hi + 2(j-4) (+1)
        const int krelA = (j < 4) ? (8 * hi + 2 * j) : (16 + 8 * hi + 2 * (j - 4));
        const int ka    = kk0 + krelA;
        au[2 * j]     = s_h1[wv][m][w0l + ka];
        au[2 * j + 1] = s_h1[wv][m][w0l + ka + 1];
        // B 32x16 bf16 layout: same K order as A (VGPR j -> K = krelA (+1)), N = m
        const int kb = kk0 + krelA;
        bu[2 * j]     = s_costT[w0l + m][16 + kb - m];
        bu[2 * j + 1] = s_costT[w0l + m][16 + kb + 1 - m];
      }
      union { v16u u; v16bf b; } A, Bv;
      A.u = au; Bv.u = bu;
      acc = __builtin_amdgcn_wmma_f32_16x16x32_bf16(
          /*neg_a=*/false, A.b, /*neg_b=*/false, Bv.b,
          /*c_mod=*/(short)0, acc, /*reuse_a=*/false, /*reuse_b=*/false);
    }

    // C/D f32 layout: VGPR r holds row M = r + 8*hi, column N = m -> coalesced rows
#pragma unroll
    for (int r = 0; r < 8; ++r) {
      const int c = r + 8 * hi;
      obase[(size_t)c * (H_ * W_) + (wb + w0l + m)] = acc[r];
    }
  }
}

extern "C" void kernel_launch(void* const* d_in, const int* in_sizes, int n_in,
                              void* d_out, int out_size, void* d_ws, size_t ws_size,
                              hipStream_t stream) {
  const float* h1   = (const float*)d_in[0];
  const float* cost = (const float*)d_in[1];
  float* out        = (float*)d_out;
  dim3 grid(B_ * H_, W_ / WBLK);   // 1024 x 4 workgroups
  dense_warp_34565896798773_kernel<<<grid, 256, 0, stream>>>(h1, cost, out);
}
